// Decoder_12532714569801
// MI455X (gfx1250) — matrix operand order repaired and checked by reading, not compiled, not run on hardware
//
#include <hip/hip_runtime.h>

// ---------------------------------------------------------------------------
// Bidirectional LSTM decoder for MI455X (gfx1250, wave32, WMMA).
//   B=64, D2=1024, H=1024, V=32000, T=100
// ---------------------------------------------------------------------------

#define BB   64
#define HH   1024
#define G4H  4096
#define VV   32000
#define TT   100
#define H2   2048

typedef __attribute__((ext_vector_type(16))) __bf16 v16bf;
typedef __attribute__((ext_vector_type(8)))  float  v8f;

union Frag16 {
    v16bf  v;
    __bf16 e[16];
};

__device__ __forceinline__ void load8(__bf16* dst, const __bf16* src) {
    *reinterpret_cast<uint4*>(dst) = *reinterpret_cast<const uint4*>(src);
}

// ---- fast transcendentals on the gfx1250 TRANS pipe -----------------------
__device__ __forceinline__ float fast_tanh(float x) {
#if __has_builtin(__builtin_amdgcn_tanhf)
    return __builtin_amdgcn_tanhf(x);          // v_tanh_f32
#else
    return tanhf(x);
#endif
}
__device__ __forceinline__ float fast_sigmoid(float x) {
    return 0.5f * fast_tanh(0.5f * x) + 0.5f;
}
__device__ __forceinline__ float fast_exp(float x) {   // x <= 0 in our uses
#if __has_builtin(__builtin_amdgcn_exp2f)
    return __builtin_amdgcn_exp2f(x * 1.4426950408889634f);  // v_exp_f32
#else
    return __expf(x);
#endif
}
__device__ __forceinline__ float fast_log(float x) {   // x >= 1 in our uses
#if __has_builtin(__builtin_amdgcn_logf)
    return __builtin_amdgcn_logf(x) * 0.6931471805599453f;   // v_log_f32
#else
    return logf(x);
#endif
}

// ---------------------------------------------------------------------------
// fp32 -> bf16 conversion (grid-stride)
// ---------------------------------------------------------------------------
__global__ void cvt_bf16_kernel(const float* __restrict__ src,
                                __bf16* __restrict__ dst, long n) {
    long i = (long)blockIdx.x * blockDim.x + threadIdx.x;
    long stride = (long)gridDim.x * blockDim.x;
    for (; i < n; i += stride) dst[i] = (__bf16)src[i];
}

// ---------------------------------------------------------------------------
// gx[k, b, g] = dot(x[b,:], W_ih[k,g,:]) + b_ih[k,g] + b_hh[k,g]   (one time)
// ---------------------------------------------------------------------------
__global__ void gx_kernel(const float* __restrict__ x,
                          const float* __restrict__ W_ih,
                          const float* __restrict__ b_ih,
                          const float* __restrict__ b_hh,
                          float* __restrict__ gx) {
    int idx = blockIdx.x * blockDim.x + threadIdx.x;  // [2*64*4096]
    int g = idx & (G4H - 1);
    int b = (idx >> 12) & (BB - 1);
    int k = idx >> 18;
    const float4* xr = (const float4*)(x + (size_t)b * HH);
    const float4* wr = (const float4*)(W_ih + ((size_t)k * G4H + g) * HH);
    float s = 0.f;
#pragma unroll 4
    for (int d = 0; d < HH / 4; ++d) {
        float4 a = xr[d], w = wr[d];
        s += a.x * w.x + a.y * w.y + a.z * w.z + a.w * w.w;
    }
    gx[idx] = s + b_ih[k * G4H + g] + b_hh[k * G4H + g];
}

// ---------------------------------------------------------------------------
// One LSTM step, fully fused: gates = gx + h @ W_hh^T via WMMA bf16,
// then i/f/g/o nonlinearities, c/h update, bf16 h for next step, bf16 out row.
// One wave (32 threads) per (dir, 16-wide h tile, 16-wide batch tile).
// grid = (H/16, B/16, 2), block = 32.
// ---------------------------------------------------------------------------
__global__ void lstm_step_kernel(const __bf16* __restrict__ h_src,
                                 __bf16* __restrict__ h_dst,
                                 float*  __restrict__ c,
                                 const float*  __restrict__ gx,
                                 const __bf16* __restrict__ Whh,
                                 __bf16* __restrict__ outs,
                                 int t) {
    const int lane = threadIdx.x;
    const int row  = lane & 15;   // also N index for B/C/D fragments
    const int hl   = lane >> 4;
    const int h0   = blockIdx.x * 16;
    const int m0   = blockIdx.y * 16;
    const int k    = blockIdx.z;

    const __bf16* A  = h_src + (size_t)k * BB * HH;        // [64, 1024]
    const __bf16* W  = Whh   + (size_t)k * G4H * HH;       // [4096, 1024]
    const float*  gk = gx    + (size_t)k * BB * G4H;       // [64, 4096]

    v8f acc[4];
    // Seed accumulators with the precomputed input/bias contribution.
    // C/D layout: VGPR r, lane l -> M = r + 8*(l>=16), N = l&15.
#pragma unroll
    for (int g = 0; g < 4; ++g) {
#pragma unroll
        for (int r = 0; r < 8; ++r) {
            int b = m0 + r + 8 * hl;
            acc[g][r] = gk[(size_t)b * G4H + g * HH + h0 + row];
        }
    }

    for (int kb = 0; kb < HH; kb += 32) {
        // A fragment 16x32 (M x K): lane (row, hl):
        //   e[0..7]  = A[row, kb + hl*8 .. +7]
        //   e[8..15] = A[row, kb + 16 + hl*8 .. +7]
        Frag16 af;
        const __bf16* ap = A + (size_t)(m0 + row) * HH + kb + hl * 8;
        load8(&af.e[0], ap);
        load8(&af.e[8], ap + 16);
#pragma unroll
        for (int g = 0; g < 4; ++g) {
            // B fragment 32x16 (K x N): lane holds col n=row, K = kb+hl*8..+7, kb+16+hl*8..+7
            Frag16 bfr;
            const __bf16* bp =
                W + (size_t)(g * HH + h0 + row) * HH + kb + hl * 8;
            load8(&bfr.e[0], bp);
            load8(&bfr.e[8], bp + 16);
            acc[g] = __builtin_amdgcn_wmma_f32_16x16x32_bf16(
                false, af.v, false, bfr.v, (short)0, acc[g], false, false);
        }
    }

    // Elementwise LSTM cell update (native v_tanh_f32 on CDNA5).
#pragma unroll
    for (int r = 0; r < 8; ++r) {
        int b = m0 + r + 8 * hl;
        int h = h0 + row;
        float iv = fast_sigmoid(acc[0][r]);
        float fv = fast_sigmoid(acc[1][r]);
        float gv = fast_tanh(acc[2][r]);
        float ov = fast_sigmoid(acc[3][r]);
        size_t ci = ((size_t)k * BB + b) * HH + h;
        float cnew = fv * c[ci] + iv * gv;
        float hnew = ov * fast_tanh(cnew);
        c[ci] = cnew;
        h_dst[ci] = (__bf16)hnew;
        outs[((size_t)b * TT + t) * H2 + k * HH + h] = (__bf16)hnew;
    }
}

// ---------------------------------------------------------------------------
// FC GEMM: logits[m, v] = outs[m, :] . fc_W[v, :] + fc_b[v]
//   M = B*T = 6400, N = V = 32000, K = 2H = 2048, bf16 x bf16 -> fp32.
// Block = 128 threads (4 waves). Block tile: 64 (M) x 256 (N); each wave owns
// a 64x64 tile (16 f32 accumulator fragments) so every B fragment is reused
// across 4 M-subtiles. A tile (64x64 bf16, 8 KB) staged through LDS per
// 64-wide K chunk. grid = (M/64 = 100, N/256 = 125).
// ---------------------------------------------------------------------------
__global__ void fc_gemm_kernel(const __bf16* __restrict__ Aouts,
                               const __bf16* __restrict__ fcW,
                               const float*  __restrict__ fc_b,
                               float* __restrict__ out) {
    __shared__ __attribute__((aligned(16))) __bf16 As[64 * 64];

    const int tid  = threadIdx.x;
    const int wave = tid >> 5;
    const int lane = tid & 31;
    const int row  = lane & 15;
    const int hl   = lane >> 4;
    const int m0   = blockIdx.x * 64;
    const int n0   = blockIdx.y * 256 + wave * 64;

    v8f acc[4][4];  // [m-subtile][n-subtile]
#pragma unroll
    for (int mi = 0; mi < 4; ++mi)
#pragma unroll
        for (int s = 0; s < 4; ++s) acc[mi][s] = (v8f){0,0,0,0,0,0,0,0};

    for (int kc = 0; kc < H2; kc += 64) {
        // Stage A[m0..m0+63, kc..kc+63] into LDS: 128 threads x 32 bf16 each.
        {
            int ar = tid >> 1;
            int ak = (tid & 1) * 32;
            const __bf16* src = Aouts + (size_t)(m0 + ar) * H2 + kc + ak;
            __bf16* dst = &As[ar * 64 + ak];
            load8(dst,      src);
            load8(dst + 8,  src + 8);
            load8(dst + 16, src + 16);
            load8(dst + 24, src + 24);
        }
        __syncthreads();

        if (kc + 64 < H2)  // hint next B chunk toward the caches
            __builtin_prefetch(fcW + (size_t)(n0 + row) * H2 + kc + 64, 0, 0);

#pragma unroll
        for (int kk = 0; kk < 64; kk += 32) {
            Frag16 af[4];
#pragma unroll
            for (int mi = 0; mi < 4; ++mi) {
                const __bf16* ap = &As[(mi * 16 + row) * 64 + kk + hl * 8];
                load8(&af[mi].e[0], ap);
                load8(&af[mi].e[8], ap + 16);
            }
#pragma unroll
            for (int s = 0; s < 4; ++s) {
                Frag16 bfr;
                const __bf16* bp =
                    fcW + (size_t)(n0 + s * 16 + row) * H2 + kc + kk + hl * 8;
                load8(&bfr.e[0], bp);
                load8(&bfr.e[8], bp + 16);
#pragma unroll
                for (int mi = 0; mi < 4; ++mi)
                    acc[mi][s] = __builtin_amdgcn_wmma_f32_16x16x32_bf16(
                        false, af[mi].v, false, bfr.v, (short)0, acc[mi][s],
                        false, false);
            }
        }
        __syncthreads();
    }

#pragma unroll
    for (int s = 0; s < 4; ++s) {
        int col = n0 + s * 16 + row;
        float bias = fc_b[col];
#pragma unroll
        for (int mi = 0; mi < 4; ++mi) {
#pragma unroll
            for (int r = 0; r < 8; ++r) {
                int m = m0 + mi * 16 + r + 8 * hl;
                out[(size_t)m * VV + col] = acc[mi][s][r] + bias;
            }
        }
    }
}

// ---------------------------------------------------------------------------
// In-place log_softmax over V per row. One 256-thread block per row.
// ---------------------------------------------------------------------------
__global__ void log_softmax_kernel(float* __restrict__ out) {
    float* p = out + (size_t)blockIdx.x * VV;
    const int tid = threadIdx.x;

    float m = -INFINITY, s = 0.f;
    for (int i = tid; i < VV; i += 256) {
        float x = p[i];
        if (x > m) { s = s * fast_exp(m - x) + 1.0f; m = x; }
        else       { s += fast_exp(x - m); }
    }
    // wave32 reduction
#pragma unroll
    for (int off = 16; off > 0; off >>= 1) {
        float m2 = __shfl_xor(m, off, 32);
        float s2 = __shfl_xor(s, off, 32);
        float mm = fmaxf(m, m2);
        s = s * fast_exp(m - mm) + s2 * fast_exp(m2 - mm);
        m = mm;
    }
    __shared__ float sm[8], ss[8];
    if ((tid & 31) == 0) { sm[tid >> 5] = m; ss[tid >> 5] = s; }
    __syncthreads();
    if (tid == 0) {
        float M = sm[0], S = ss[0];
#pragma unroll
        for (int i = 1; i < 8; ++i) {
            float mm = fmaxf(M, sm[i]);
            S = S * fast_exp(M - mm) + ss[i] * fast_exp(sm[i] - mm);
            M = mm;
        }
        sm[0] = M;
        ss[0] = fast_log(S);
    }
    __syncthreads();
    float lse = sm[0] + ss[0];
    for (int i = tid; i < VV; i += 256) p[i] -= lse;
}

// ---------------------------------------------------------------------------
// Host-side launcher
// ---------------------------------------------------------------------------
extern "C" void kernel_launch(void* const* d_in, const int* in_sizes, int n_in,
                              void* d_out, int out_size, void* d_ws,
                              size_t ws_size, hipStream_t stream) {
    (void)in_sizes; (void)n_in; (void)out_size; (void)ws_size;

    const float* x    = (const float*)d_in[0];
    const float* h0   = (const float*)d_in[1];
    const float* c0   = (const float*)d_in[2];
    const float* W_ih = (const float*)d_in[3];
    const float* W_hh = (const float*)d_in[4];
    const float* b_ih = (const float*)d_in[5];
    const float* b_hh = (const float*)d_in[6];
    const float* fc_W = (const float*)d_in[7];
    const float* fc_b = (const float*)d_in[8];
    float* out = (float*)d_out;

    // Workspace layout (all comfortably aligned)
    char* ws = (char*)d_ws;
    size_t off = 0;
    __bf16* Whh_bf = (__bf16*)(ws + off); off += (size_t)2 * G4H * HH * 2;      // 16 MB
    __bf16* fcW_bf = (__bf16*)(ws + off); off += (size_t)VV * H2 * 2;           // 131 MB
    float*  gx     = (float*) (ws + off); off += (size_t)2 * BB * G4H * 4;      // 2 MB
    __bf16* h_a    = (__bf16*)(ws + off); off += (size_t)2 * BB * HH * 2;       // 256 KB
    __bf16* h_b    = (__bf16*)(ws + off); off += (size_t)2 * BB * HH * 2;       // 256 KB
    float*  cbuf   = (float*) (ws + off); off += (size_t)2 * BB * HH * 4;       // 512 KB
    __bf16* outs   = (__bf16*)(ws + off); off += (size_t)BB * TT * H2 * 2;      // 26 MB

    // 1) One-time conversions to bf16 / fp32 working state.
    cvt_bf16_kernel<<<4096, 256, 0, stream>>>(W_hh, Whh_bf, (long)2 * G4H * HH);
    cvt_bf16_kernel<<<8192, 256, 0, stream>>>(fc_W, fcW_bf, (long)VV * H2);
    cvt_bf16_kernel<<<512, 256, 0, stream>>>(h0, h_a, (long)2 * BB * HH);
    hipMemcpyAsync(cbuf, c0, (size_t)2 * BB * HH * sizeof(float),
                   hipMemcpyDeviceToDevice, stream);

    // 2) Input-gate precompute (x is constant over time).
    gx_kernel<<<(2 * BB * G4H) / 256, 256, 0, stream>>>(x, W_ih, b_ih, b_hh, gx);

    // 3) 100 fused recurrent steps, ping-ponging the bf16 hidden state.
    for (int t = 0; t < TT; ++t) {
        const __bf16* hs = (t & 1) ? h_b : h_a;
        __bf16*       hd = (t & 1) ? h_a : h_b;
        lstm_step_kernel<<<dim3(HH / 16, BB / 16, 2), 32, 0, stream>>>(
            hs, hd, cbuf, gx, Whh_bf, outs, t);
    }

    // 4) Vocab projection (dominant GEMM) + bias.
    fc_gemm_kernel<<<dim3((BB * TT) / 64, VV / 256), 128, 0, stream>>>(
        outs, fcW_bf, fc_b, out);

    // 5) In-place log_softmax per (b, t) row.
    log_softmax_kernel<<<BB * TT, 256, 0, stream>>>(out);
}
